// SpatioTemporalEmbLoss_87436944212099
// MI455X (gfx1250) — matrix operand order repaired and checked by reading, not compiled, not run on hardware
//
#include <hip/hip_runtime.h>
#include <hip/hip_bf16.h>

// ---------- types ----------
typedef __attribute__((ext_vector_type(16))) _Float16 v16h;
typedef __attribute__((ext_vector_type(8)))  float    v8f;
typedef __attribute__((ext_vector_type(4)))  unsigned int u32x4;
typedef __attribute__((ext_vector_type(8)))  int      i32x8;
typedef __attribute__((ext_vector_type(4)))  int      i32x4;

// ---------- problem constants ----------
#define T_  8
#define H_  480
#define W_  480
#define SP  (T_*H_*W_)        // 1,843,200 pixels per batch
#define NB  16384             // lovasz error histogram bins over [0,2]
#define TILE 1024             // pixels per TDM tile (SP % TILE == 0 -> 1800 tiles)

// ---------- workspace layout (float words) ----------
#define ACC_OFF  0                       // acc[2][16][16]   (feature x instance sums)
#define PAR_OFF  512                     // par[2][9][8]
#define LOV_OFF  656                     // lov[2][9]
#define SFG_OFF  674                     // sfg[2][9]
#define HPOS_OFF 1024                    // uint  [16][NB]
#define HNEG_OFF (HPOS_OFF + 16*NB)      // uint  [16][NB]
#define HE_OFF   (HPOS_OFF + 32*NB)      // float [16][NB]
#define WS_WORDS (HPOS_OFF + 48*NB)      // ~3.0 MB

// fast transcendentals: v_rcp_f32 instead of IEEE divide sequence
__device__ __forceinline__ float rcpf_(float v){ return __builtin_amdgcn_rcpf(v); }
__device__ __forceinline__ float sigmoidf_(float v){ return rcpf_(1.f + __expf(-v)); }
__device__ __forceinline__ float tanhf_(float v){ return 1.f - 2.f*rcpf_(__expf(2.f*v) + 1.f); }

// ---------- zero workspace ----------
__global__ void kinit(unsigned int* w, int n){
  for (int i = blockIdx.x*blockDim.x + threadIdx.x; i < n; i += gridDim.x*blockDim.x) w[i] = 0u;
}

// ---------- pass 1: WMMA segmented reduction -> acc[b][feature][instance] ----------
// features: 0=1(count) 1=x 2=y 3=z 4=sigma 5=sigma^2 6=seed^2 (col 0 of feat 6 = seed_bg)
__global__ void pass1(const float* __restrict__ pred, const long long* __restrict__ inst,
                      float* __restrict__ acc) {
  const int b    = blockIdx.y;
  const int tid  = threadIdx.x;
  const int lane = tid & 31;
  const int wave = tid >> 5;
  const bool lo  = lane < 16;
  const int  m   = lane & 15;

  __shared__ _Float16 fLDS[8][32][16];
  __shared__ int      iLDS[8][32];

  const float*     sig = pred + ((size_t)b*5 + 3) * SP;
  const float*     sed = pred + ((size_t)b*5 + 4) * SP;
  const long long* ins = inst + (size_t)b * SP;

  v8f c = {};
  const int stride = gridDim.x * blockDim.x;
  const int iters  = (SP + stride - 1) / stride;

  for (int it = 0; it < iters; ++it) {
    int p = it*stride + blockIdx.x*blockDim.x + tid;
    bool valid = p < SP;
    int pc = valid ? p : 0;
    int t  = pc / (H_*W_);
    int rm = pc - t*(H_*W_);
    int y  = rm / W_;
    int x  = rm - y*W_;
    float vf = valid ? 1.f : 0.f;
    float s  = sig[pc] * vf;
    float sd = sigmoidf_(sed[pc]);
    float fv[16];
    fv[0] = vf;
    fv[1] = vf * (float)x * (1.f/(W_-1));
    fv[2] = vf * (float)y * (1.f/(H_-1));
    fv[3] = vf * (float)t * (0.15f/(T_-1));
    fv[4] = s;
    fv[5] = s*s;
    fv[6] = vf * sd*sd;
    #pragma unroll
    for (int j = 7; j < 16; ++j) fv[j] = 0.f;
    #pragma unroll
    for (int j = 0; j < 16; ++j) fLDS[wave][lane][j] = (_Float16)fv[j];
    iLDS[wave][lane] = valid ? (int)ins[pc] : -1;
    __syncthreads();

    // A: 16x32 f16 (rows=features).  lane<16: K in {0..7,16..23}; lane>=16: K in {8..15,24..31}
    // B: 32x16 f16 one-hot (cols=instance id). lane<16: K in {0..7,16..23}; lane>=16: K in {8..15,24..31}
    v16h a, bm;
    #pragma unroll
    for (int j = 0; j < 16; ++j) {
      int kA = lo ? ((j < 8) ? j : j + 8) : ((j < 8) ? j + 8 : j + 16);
      a[j]  = fLDS[wave][kA][m];
      int kB = kA;
      bm[j] = (iLDS[wave][kB] == m) ? (_Float16)1.0f : (_Float16)0.0f;
    }
    c = __builtin_amdgcn_wmma_f32_16x16x32_f16(false, a, false, bm, (short)0, c, false, false);
    __syncthreads();
  }

  // C/D layout: VGPR v -> (M=v, N=lane) lanes 0-15; (M=v+8, N=lane-16) lanes 16-31
  #pragma unroll
  for (int v = 0; v < 8; ++v) {
    int M = lo ? v : v + 8;
    atomicAdd(&acc[(size_t)b*256 + M*16 + m], c[v]);
  }
}

// ---------- per-instance parameters ----------
__global__ void kparams(const float* __restrict__ acc, float* __restrict__ par) {
  int t = threadIdx.x;
  if (t < 16) {
    int b = t >> 3, k = (t & 7) + 1;
    const float* A = acc + b*256;
    float cnt  = A[0*16 + k];
    float safe = fmaxf(cnt, 1.f);
    float rs   = rcpf_(safe);
    float present = cnt > 0.f ? 1.f : 0.f;
    float cx = A[1*16 + k] * rs;
    float cy = A[2*16 + k] * rs;
    float cz = A[3*16 + k] * rs;
    float sm = A[4*16 + k] * rs;
    float s2 = A[5*16 + k] * rs;
    float var = fmaxf(s2 - sm*sm, 0.f) * present;   // mean((s-mean)^2)
    float* P = par + (b*9 + k)*8;
    P[0]=cx; P[1]=cy; P[2]=cz; P[3]=__expf(sm*10.f); P[4]=var; P[5]=present; P[6]=cnt; P[7]=0.f;
  }
  if (t < 2) par[(t*9)*8] = acc[t*256 + 6*16 + 0];  // seed_bg
}

// ---------- pass 2: dual TDM tile load (pred 5ch + int64 instance) + histogram ----------
__global__ void pass2(const float* __restrict__ pred, const long long* __restrict__ inst,
                      const float* __restrict__ par,
                      unsigned int* __restrict__ hpos, unsigned int* __restrict__ hneg,
                      float* __restrict__ he, float* __restrict__ sfg) {
  const int b = blockIdx.y;
  __shared__ float     tileF[5][TILE];   // 20 KB, TDM-filled
  __shared__ long long tileI[TILE];      //  8 KB, TDM-filled

  float cx[8], cy[8], cz[8], sx[8]; int pres[8];
  #pragma unroll
  for (int k = 0; k < 8; ++k) {
    const float* P = par + (b*9 + k + 1)*8;
    cx[k]=P[0]; cy[k]=P[1]; cz[k]=P[2]; sx[k]=P[3]; pres[k] = P[5] > 0.f;
  }
  const long long* ibase = inst + (size_t)b*SP;
  const float*     pbase = pred + (size_t)b*5*SP;
  const int nt = SP / TILE;  // 1800

  for (int tile = blockIdx.x; tile < nt; tile += gridDim.x) {
    const int start = tile * TILE;
    if (threadIdx.x < 32) {
      i32x4 z4 = {0,0,0,0};
      i32x8 z8 = {0,0,0,0,0,0,0,0};
      // ---- descriptor 1: prediction tile (TILE pixels x 5 channels, 4B, row stride SP) ----
      {
        unsigned long long ga = (unsigned long long)(const void*)(pbase + start);
        unsigned int ldsa = (unsigned int)(unsigned long long)(void*)&tileF[0][0];
        u32x4 g0;
        g0[0] = 1u;
        g0[1] = ldsa;
        g0[2] = (unsigned int)ga;
        g0[3] = ((unsigned int)(ga >> 32) & 0x01FFFFFFu) | 0x80000000u; // addr hi | type=2
        i32x8 g1;
        g1[0] = 0x00020000;                                 // data_size=4B
        g1[1] = (int)((SP & 0xFFFF) << 16);                 // tensor_dim0 lo16
        g1[2] = (int)((SP >> 16) & 0xFFFF) | (5 << 16);     // tensor_dim0 hi16 | tensor_dim1=5
        g1[3] = (int)(TILE << 16);                          // tile_dim0
        g1[4] = 5;                                          // tile_dim1=5
        g1[5] = (int)SP;                                    // tensor_dim0_stride
        g1[6] = 0; g1[7] = 0;
        __builtin_amdgcn_tensor_load_to_lds(g0, g1, z4, z4, z8, 0);
      }
      // ---- descriptor 2: instance tile (TILE x 1, 8B int64) ----
      {
        unsigned long long ga = (unsigned long long)(const void*)(ibase + start);
        unsigned int ldsa = (unsigned int)(unsigned long long)(void*)&tileI[0];
        u32x4 g0;
        g0[0] = 1u;
        g0[1] = ldsa;
        g0[2] = (unsigned int)ga;
        g0[3] = ((unsigned int)(ga >> 32) & 0x01FFFFFFu) | 0x80000000u;
        i32x8 g1;
        g1[0] = 0x00030000;                                 // data_size=8B
        g1[1] = (int)((SP & 0xFFFF) << 16);                 // tensor_dim0 lo16
        g1[2] = (int)((SP >> 16) & 0xFFFF) | (1 << 16);     // tensor_dim0 hi16 | tensor_dim1=1
        g1[3] = (int)(TILE << 16);                          // tile_dim0
        g1[4] = 1;                                          // tile_dim1=1
        g1[5] = (int)SP;
        g1[6] = 0; g1[7] = 0;
        __builtin_amdgcn_tensor_load_to_lds(g0, g1, z4, z4, z8, 0);
      }
      __builtin_amdgcn_s_wait_tensorcnt(0);
    }
    __syncthreads();

    int next = tile + gridDim.x;
    if (next < nt)
      __builtin_prefetch((const void*)(pbase + next*TILE + threadIdx.x*16), 0, 3);

    for (int i = threadIdx.x; i < TILE; i += blockDim.x) {
      int p  = start + i;
      int t  = p / (H_*W_);
      int rm = p - t*(H_*W_);
      int y  = rm / W_;
      int x  = rm - y*W_;
      float e0 = tanhf_(tileF[0][i]) + (float)x * (1.f/(W_-1));
      float e1 = tanhf_(tileF[1][i]) + (float)y * (1.f/(H_-1));
      float e2 = tanhf_(tileF[2][i]) + (float)t * (0.15f/(T_-1));
      float seed = sigmoidf_(tileF[4][i]);
      int id = (int)tileI[i];
      #pragma unroll
      for (int k = 0; k < 8; ++k) {
        if (!pres[k]) continue;
        float dx = e0 - cx[k], dy = e1 - cy[k], dz = e2 - cz[k];
        float dist = __expf(-(dx*dx + dy*dy + dz*dz) * sx[k]);
        bool fg = (id == k + 1);
        float err = fg ? (2.f - 2.f*dist) : (2.f*dist);   // errors >= 0 -> relu is identity
        int bin = (int)(err * (NB * 0.5f));
        bin = bin < 0 ? 0 : (bin > NB-1 ? NB-1 : bin);
        int idx = (b*8 + k)*NB + bin;
        if (fg) atomicAdd(&hpos[idx], 1u); else atomicAdd(&hneg[idx], 1u);
        atomicAdd(&he[idx], err);
        if (fg) { float d = seed - dist; atomicAdd(&sfg[b*9 + k + 1], d*d); }
      }
    }
    __syncthreads();
  }
}

// ---------- pass 3: Lovasz hinge from histograms (telescoped jaccard gradient) ----------
__global__ void pass3(const unsigned int* __restrict__ hpos, const unsigned int* __restrict__ hneg,
                      const float* __restrict__ he, const float* __restrict__ par,
                      float* __restrict__ lov) {
  const int combo = blockIdx.x;            // 0..15
  const int b = combo >> 3, k = combo & 7;
  const float* P = par + (b*9 + k + 1)*8;
  const float present = P[5];
  const float G = P[6];                    // instance pixel count
  const int t = threadIdx.x;
  __shared__ float sPos[256], sTot[256], sLoss[256];
  const unsigned int* hp = hpos + (size_t)combo*NB;
  const unsigned int* hn = hneg + (size_t)combo*NB;
  const float*        hE = he   + (size_t)combo*NB;
  const int per = NB / 256;                // 64 bins/thread
  const int lo = t*per, hi = lo + per;

  float posL = 0.f, totL = 0.f;
  for (int i = lo; i < hi; ++i) { unsigned p = hp[i], n = hn[i]; posL += (float)p; totL += (float)(p+n); }
  sPos[t] = posL; sTot[t] = totL;
  __syncthreads();
  float cumG = 0.f, cumT = 0.f;            // suffix: elements with larger error
  for (int u = t + 1; u < 256; ++u) { cumG += sPos[u]; cumT += sTot[u]; }

  float loss = 0.f;
  for (int i = hi - 1; i >= lo; --i) {     // descending error order
    unsigned p = hp[i], n = hn[i];
    float ntot = (float)(p + n);
    if (ntot > 0.f) {
      float meanE = hE[i] * rcpf_(ntot);
      float jb = 1.f - (G - cumG) * rcpf_(G + cumT - cumG);
      cumG += (float)p; cumT += ntot;
      float ja = 1.f - (G - cumG) * rcpf_(G + cumT - cumG);
      loss += meanE * (ja - jb);
    }
  }
  sLoss[t] = loss;
  __syncthreads();
  for (int s = 128; s > 0; s >>= 1) { if (t < s) sLoss[t] += sLoss[t + s]; __syncthreads(); }
  if (t == 0) lov[b*9 + k + 1] = (present > 0.f) ? sLoss[0] : 0.f;
}

// ---------- finalize ----------
__global__ void kfinal(const float* __restrict__ par, const float* __restrict__ lov,
                       const float* __restrict__ sfg, float* __restrict__ out) {
  if (threadIdx.x == 0 && blockIdx.x == 0) {
    float total = 0.f;
    for (int b = 0; b < 2; ++b) {
      float pres = 0.f, lsum = 0.f, vsum = 0.f, ssum = 0.f;
      for (int k = 1; k <= 8; ++k) {
        const float* P = par + (b*9 + k)*8;
        pres += P[5]; vsum += P[4]; lsum += lov[b*9 + k]; ssum += sfg[b*9 + k];
      }
      float denom = fmaxf(pres, 1.f);
      float seed_bg = par[(b*9)*8];
      total += lsum/denom + 10.f*(vsum/denom) + (seed_bg + ssum) * (1.f/(float)(H_*W_));
    }
    out[0] = total * 0.5f;
  }
}

extern "C" void kernel_launch(void* const* d_in, const int* in_sizes, int n_in,
                              void* d_out, int out_size, void* d_ws, size_t ws_size,
                              hipStream_t stream) {
  const float*     pred = (const float*)d_in[0];
  const long long* inst = (const long long*)d_in[1];   // int64 instance ids (values 0..8)
  float* ws  = (float*)d_ws;
  float* acc = ws + ACC_OFF;
  float* par = ws + PAR_OFF;
  float* lov = ws + LOV_OFF;
  float* sfg = ws + SFG_OFF;
  unsigned int* hpos = (unsigned int*)(ws + HPOS_OFF);
  unsigned int* hneg = (unsigned int*)(ws + HNEG_OFF);
  float* he = ws + HE_OFF;

  kinit<<<1024, 256, 0, stream>>>((unsigned int*)ws, WS_WORDS);
  dim3 g1(1024, 2);
  pass1<<<g1, 256, 0, stream>>>(pred, inst, acc);
  kparams<<<1, 64, 0, stream>>>(acc, par);
  dim3 g2(256, 2);
  pass2<<<g2, 256, 0, stream>>>(pred, inst, par, hpos, hneg, he, sfg);
  pass3<<<16, 256, 0, stream>>>(hpos, hneg, he, par, lov);
  kfinal<<<1, 32, 0, stream>>>(par, lov, sfg, (float*)d_out);
}
